// QMixer_43276090475091
// MI455X (gfx1250) — matrix operand order repaired and checked by reading, not compiled, not run on hardware
//
#include <hip/hip_runtime.h>
#include <hip/hip_bf16.h>

// ---------------------------------------------------------------------------
// QMixer (GAT + hypernet mixing) fused kernel for gfx1250 (MI455X, wave32).
//   B = 64*256 = 16384 rows, G=8 agents, HID=128, SDIM=256, EMB=32.
// One workgroup = 256 threads (8 wave32) handles BTILE=16 batch rows.
// All three GEMMs run on v_wmma_f32_16x16x32_f16.
// The obs tile (64 KB, the dominant HBM stream) is DMA'd into LDS by the
// Tensor Data Mover (tensor_load_to_lds) overlapped with gat_W staging.
// ---------------------------------------------------------------------------

typedef _Float16 f16;
typedef __attribute__((ext_vector_type(8)))  _Float16 v8h;
typedef __attribute__((ext_vector_type(16))) _Float16 v16h;
typedef __attribute__((ext_vector_type(8)))  float    v8f;
typedef __attribute__((ext_vector_type(4)))  unsigned int u32x4;
typedef __attribute__((ext_vector_type(8)))  int i32x8;
typedef __attribute__((ext_vector_type(4)))  int i32x4;

#define BTILE 16
#define NBATCH 16384

// LDS layout (bytes). Regions R1/R2/H are reused across phases.
#define OFF_A     0        // f16 [128][136]  GAT A tile; later hidden f16 [16][1032]
#define OFF_WT    34816    // f16 [128][136]  gat_W^T; later BT chunks [320][40] / [32][328]
#define OFF_H     69632    // TDM obs f32 staging [16384]; then f32 [128][130] GAT h
#define OFF_S     136192   // f16 [16][264]   states tile (f16)
#define OFF_E     144640   // f32 [2][128]    e_src / e_dst
#define OFF_ALPHA 145664   // f32 [16][8][8]
#define OFF_QS    149760   // f32 [16][8]
#define OFF_GB    150272   // f32 [128]
#define OFF_ASRC  150784   // f32 [128]
#define OFF_ADST  151296   // f32 [128]
#define OFF_HYP   151808   // f32 [16][324]   hypernet outputs (320 cols + pad)
#define OFF_MIX   172544   // f32 [16][32]    elu(qs@|w1|+b1) * |wf|
#define OFF_VP    174592   // f32 [8][256]    v1 partial accumulators per wave
#define OFF_VR    182784   // f32 [16][32]    relu(v1) activations
#define SMEM_BYTES 184832

__device__ __forceinline__ v16h mk16(v8h lo, v8h hi) {
  v16h r;
#pragma unroll
  for (int i = 0; i < 8; ++i) { r[i] = lo[i]; r[i + 8] = hi[i]; }
  return r;
}
__device__ __forceinline__ v8h ld8(const f16* p) { return *(const v8h*)p; }
__device__ __forceinline__ v8f vzero() {
  v8f z;
#pragma unroll
  for (int i = 0; i < 8; ++i) z[i] = 0.0f;
  return z;
}
#define WMMA_F16(A, Bf, C) \
  __builtin_amdgcn_wmma_f32_16x16x32_f16(false, (A), false, (Bf), (short)0, (C), false, false)

extern "C" __global__ __launch_bounds__(256, 1)
void qmixer_fused(const float* __restrict__ agent_qs, const float* __restrict__ states,
                  const float* __restrict__ obs,      const float* __restrict__ gat_W,
                  const float* __restrict__ a_src,    const float* __restrict__ a_dst,
                  const float* __restrict__ gat_b,    const float* __restrict__ w1_W,
                  const float* __restrict__ w1_b,     const float* __restrict__ hb_W,
                  const float* __restrict__ hb_b,     const float* __restrict__ wf_W,
                  const float* __restrict__ wf_b,     const float* __restrict__ v1_W,
                  const float* __restrict__ v1_b,     const float* __restrict__ v2_W,
                  const float* __restrict__ v2_b,     float* __restrict__ out)
{
  extern __shared__ char smem[];
  f16*   shA   = (f16*)(smem + OFF_A);      // [128][136]
  f16*   shHid = (f16*)(smem + OFF_A);      // [16][1032] (overlays shA after GAT GEMM)
  f16*   shWT  = (f16*)(smem + OFF_WT);     // [128][136]
  f16*   shBT  = (f16*)(smem + OFF_WT);     // [320][40]  (reuse)
  f16*   shV1T = (f16*)(smem + OFF_WT);     // [32][328]  (reuse)
  float* shH   = (float*)(smem + OFF_H);    // [128][130]; first TDM obs f32 [16384]
  f16*   shS   = (f16*)(smem + OFF_S);      // [16][264]
  float* shE   = (float*)(smem + OFF_E);    // [2][128]
  float* shAl  = (float*)(smem + OFF_ALPHA);
  float* shQs  = (float*)(smem + OFF_QS);
  float* shGb  = (float*)(smem + OFF_GB);
  float* shAs  = (float*)(smem + OFF_ASRC);
  float* shAd  = (float*)(smem + OFF_ADST);
  float* shHyp = (float*)(smem + OFF_HYP);  // [16][324]
  float* shMix = (float*)(smem + OFF_MIX);
  float* shVP  = (float*)(smem + OFF_VP);   // [8][256]
  float* shVR  = (float*)(smem + OFF_VR);

  const int tid  = (int)threadIdx.x;
  const int lane = tid & 31;
  const int wv   = tid >> 5;
  const int b0   = (int)blockIdx.x * BTILE;

  // --------------------------------------------------------------------
  // Stage 0a: kick the Tensor Data Mover on the obs tile (64 KB contiguous
  // f32 -> LDS @ OFF_H). Issued by wave 0 only (TDM ignores EXEC; one
  // issuing wave avoids duplicate DMAs). D# per CDNA5 ISA
  // section 8: 1-row tile, data_size=4B, tile_dim0=tensor_dim0=16384.
  // --------------------------------------------------------------------
  if (wv == 0) {
    unsigned long long ga =
        (unsigned long long)(const void*)(obs + (size_t)b0 * (8 * 128));
    u32x4 g0;
    g0[0] = 1u;                                           // count=1 (valid user D#)
    g0[1] = (unsigned)OFF_H;                              // lds_addr (bytes)
    g0[2] = (unsigned)(ga & 0xFFFFFFFFull);               // global_addr[31:0]
    g0[3] = (unsigned)((ga >> 32) & 0x01FFFFFFull)        // global_addr[56:32]
            | 0x80000000u;                                // type=2 ("image")
    i32x8 g1;
    g1[0] = (int)(2u << 16);          // workgroup_mask=0, data_size=2 (4 bytes)
    g1[1] = (int)(16384u << 16);      // tensor_dim0[15:0] in bits[31:16]
    g1[2] = (int)(1u << 16);          // tensor_dim0[31:16]=0 | tensor_dim1=1
    g1[3] = (int)(16384u << 16);      // tensor_dim1 hi=0 | tile_dim0=16384
    g1[4] = 1;                        // tile_dim1=1, tile_dim2=0
    g1[5] = 16384;                    // tensor_dim0_stride[31:0]
    g1[6] = 0;                        // stride hi / tensor_dim1_stride lo
    g1[7] = 0;                        // tensor_dim1_stride hi
    i32x4 z4 = {0, 0, 0, 0};
#if defined(__clang_major__) && __clang_major__ >= 23
    i32x8 z8 = {0, 0, 0, 0, 0, 0, 0, 0};
    __builtin_amdgcn_tensor_load_to_lds(g0, g1, z4, z4, z8, 0);
#else
    __builtin_amdgcn_tensor_load_to_lds(g0, g1, z4, z4, 0);
#endif
  }

  // --------------------------------------------------------------------
  // Stage 0b: cooperative staging of everything else, overlapped with TDM.
  // --------------------------------------------------------------------
  for (int i = tid; i < 128 * 128; i += 256) {        // gat_W [d][e] -> shWT[e][d]
    int d = i >> 7, e = i & 127;
    shWT[e * 136 + d] = (f16)gat_W[i];
  }
  {
    const float* st = states + (size_t)b0 * 256;
    for (int i = tid; i < 16 * 256; i += 256) {
      int b = i >> 8, k = i & 255;
      shS[b * 264 + k] = (f16)st[i];
    }
  }
  if (tid < 128) {
    shQs[tid] = agent_qs[b0 * 8 + tid];
    shGb[tid] = gat_b[tid];
    shAs[tid] = a_src[tid];
    shAd[tid] = a_dst[tid];
  }
  if (wv == 0) __builtin_amdgcn_s_wait_tensorcnt(0);  // publish TDM before barrier
  __syncthreads();

  // Stage 0c: convert TDM-staged obs f32 (flat in shH region) -> f16 A tile.
  {
    const float* of = (const float*)(smem + OFF_H);
    for (int i = tid; i < 128 * 128; i += 256) {
      int r = i >> 7, k = i & 127;
      shA[r * 136 + k] = (f16)of[i];
    }
  }
  __syncthreads();   // orders convert-reads of shH before GEMM-writes of shH

  // --------------------------------------------------------------------
  // Stage 1: GAT GEMM  h[128x128] = A[128x128] * gat_W  via WMMA f16
  // wave wv owns M rows [16wv, 16wv+16), all 8 N tiles.
  // --------------------------------------------------------------------
  {
    v8f acc[8];
#pragma unroll
    for (int n = 0; n < 8; ++n) acc[n] = vzero();
    const int m   = lane & 15;
    const int kgA = (lane >> 4) * 16;   // A: low lanes K 0..15, high lanes K 16..31
    const int kgB = (lane >> 4) * 16;   // B: low lanes K 0..15, high lanes K 16..31
#pragma unroll
    for (int ks = 0; ks < 4; ++ks) {
      const int kb = ks * 32;
      const f16* ap = shA + (wv * 16 + m) * 136 + kb + kgA;
      v16h afr = mk16(ld8(ap), ld8(ap + 8));
#pragma unroll
      for (int nt = 0; nt < 8; ++nt) {
        const f16* bp = shWT + (nt * 16 + m) * 136 + kb + kgB;
        v16h bfr = mk16(ld8(bp), ld8(bp + 8));
        acc[nt] = WMMA_F16(afr, bfr, acc[nt]);
      }
    }
    const int rowb = wv * 16 + (lane >> 4) * 8;
#pragma unroll
    for (int nt = 0; nt < 8; ++nt)
#pragma unroll
      for (int r = 0; r < 8; ++r)
        shH[(rowb + r) * 130 + nt * 16 + m] = acc[nt][r];
  }
  __syncthreads();

  // --------------------------------------------------------------------
  // Stage 2: e_src/e_dst dot products (256 length-128 dots, one per thread)
  // --------------------------------------------------------------------
  {
    int row = tid & 127;
    const float* vec = (tid < 128) ? shAs : shAd;
    const float2* hp2 = (const float2*)(shH + row * 130);   // 520 B rows: 8B aligned
    const float2* vc2 = (const float2*)vec;
    float s = 0.f;
    for (int k = 0; k < 64; ++k) {
      float2 h2 = hp2[k], w2 = vc2[k];
      s += h2.x * w2.x + h2.y * w2.y;
    }
    shE[(tid >> 7) * 128 + row] = s;   // shE[0][.]=e_src, shE[1][.]=e_dst
  }
  __syncthreads();

  // --------------------------------------------------------------------
  // Stage 3: leaky-relu logits, self-masked softmax -> alpha [16][8][8]
  // --------------------------------------------------------------------
  if (tid < 128) {
    int b = tid >> 3, i = tid & 7;
    float ed = shE[128 + b * 8 + i];
    float lg[8], mx = -1e30f;
#pragma unroll
    for (int j = 0; j < 8; ++j) {
      if (j == i) { lg[j] = -1e30f; continue; }
      float x = ed + shE[b * 8 + j];
      x = (x >= 0.f) ? x : 0.2f * x;
      lg[j] = x; mx = fmaxf(mx, x);
    }
    float se = 0.f;
#pragma unroll
    for (int j = 0; j < 8; ++j) {
      float e = (j == i) ? 0.f : __expf(lg[j] - mx);
      lg[j] = e; se += e;
    }
    float inv = 1.f / se;
#pragma unroll
    for (int j = 0; j < 8; ++j) shAl[b * 64 + i * 8 + j] = lg[j] * inv;
  }
  __syncthreads();

  // --------------------------------------------------------------------
  // Stage 4: hidden = alpha @ h + gat_b -> f16 into combined buffer (overlays shA)
  // --------------------------------------------------------------------
  for (int t = 0; t < 64; ++t) {
    int idx = tid + t * 256;
    int d = idx & 127, gi = (idx >> 7) & 7, b = idx >> 10;
    const float* al = shAl + b * 64 + gi * 8;
    const float* hp = shH + (b * 8) * 130 + d;
    float s = 0.f;
#pragma unroll
    for (int j = 0; j < 8; ++j) s += al[j] * hp[j * 130];
    s += shGb[d];
    shHid[b * 1032 + gi * 128 + d] = (f16)s;
  }
  __syncthreads();

  // --------------------------------------------------------------------
  // Stage 5: hypernet GEMM [16 x 320] = s[16x256] @ [w1_W|hb_W|wf_W]
  // K staged in 8 chunks of 32 (f16, transposed). wave wv owns N tiles wv, wv+8, wv+16.
  // --------------------------------------------------------------------
  {
    v8f hacc[3];
    hacc[0] = vzero(); hacc[1] = vzero(); hacc[2] = vzero();
    const int m = lane & 15;
    for (int kc = 0; kc < 8; ++kc) {
      for (int i = tid; i < 320 * 32; i += 256) {
        int n = i % 320, kl = i / 320;
        int k = kc * 32 + kl;
        float w;
        if (n < 256)      w = w1_W[k * 256 + n];
        else if (n < 288) w = hb_W[k * 32 + (n - 256)];
        else              w = wf_W[k * 32 + (n - 288)];
        shBT[n * 40 + kl] = (f16)w;
      }
      __syncthreads();
      const f16* ap = shS + m * 264 + kc * 32 + (lane >> 4) * 16;
      v16h afr = mk16(ld8(ap), ld8(ap + 8));
      const int kgB = (lane >> 4) * 16;
      {
        const f16* bp = shBT + ((wv)      * 16 + m) * 40 + kgB;
        hacc[0] = WMMA_F16(afr, mk16(ld8(bp), ld8(bp + 8)), hacc[0]);
        bp = shBT + ((wv + 8)  * 16 + m) * 40 + kgB;
        hacc[1] = WMMA_F16(afr, mk16(ld8(bp), ld8(bp + 8)), hacc[1]);
        if (wv < 4) {
          bp = shBT + ((wv + 16) * 16 + m) * 40 + kgB;
          hacc[2] = WMMA_F16(afr, mk16(ld8(bp), ld8(bp + 8)), hacc[2]);
        }
      }
      __syncthreads();
    }
    const int br = (lane >> 4) * 8;
#pragma unroll
    for (int r = 0; r < 8; ++r) {
      shHyp[(br + r) * 324 + (wv)     * 16 + m] = hacc[0][r];
      shHyp[(br + r) * 324 + (wv + 8) * 16 + m] = hacc[1][r];
      if (wv < 4) shHyp[(br + r) * 324 + (wv + 16) * 16 + m] = hacc[2][r];
    }
  }
  __syncthreads();

  // --------------------------------------------------------------------
  // Stage 6: mixing head: elu(qs @ |w1| + b1) * |w_final| -> shMix [16][32]
  // --------------------------------------------------------------------
#pragma unroll
  for (int t = 0; t < 2; ++t) {
    int idx = tid + t * 256;
    int b = idx >> 5, e = idx & 31;
    float s = 0.f;
#pragma unroll
    for (int g = 0; g < 8; ++g)
      s += fabsf(shHyp[b * 324 + g * 32 + e] + w1_b[g * 32 + e]) * shQs[b * 8 + g];
    s += shHyp[b * 324 + 256 + e] + hb_b[e];
    float el = (s > 0.f) ? s : (__expf(s) - 1.f);
    float wf = fabsf(shHyp[b * 324 + 288 + e] + wf_b[e]);
    shMix[b * 32 + e] = el * wf;
  }

  // --------------------------------------------------------------------
  // Stage 7: V-net GEMM [16 x 32] = combined[16x1280] @ v1_W, 4-way K split.
  // wave wv: n-tile = wv&1, k-phase = wv>>1 (steps s with s%4==phase).
  // --------------------------------------------------------------------
  {
    v8f vacc = vzero();
    const int vnt = wv & 1;
    const int vkq = wv >> 1;
    const int m = lane & 15;
    for (int kc = 0; kc < 4; ++kc) {
      for (int i = tid; i < 32 * 320; i += 256) {
        int kl = i >> 5, n = i & 31;
        shV1T[n * 328 + kl] = (f16)v1_W[(size_t)(kc * 320 + kl) * 32 + n];
      }
      __syncthreads();
#pragma unroll
      for (int s = 0; s < 10; ++s) {
        if ((s & 3) != vkq) continue;              // wave-uniform
        int kglob = kc * 320 + s * 32;
        const f16* ap = (kglob < 256) ? (shS + m * 264 + kglob)
                                      : (shHid + m * 1032 + (kglob - 256));
        ap += (lane >> 4) * 16;
        v16h afr = mk16(ld8(ap), ld8(ap + 8));
        const f16* bp = shV1T + (vnt * 16 + m) * 328 + s * 32 + (lane >> 4) * 16;
        v16h bfr = mk16(ld8(bp), ld8(bp + 8));
        vacc = WMMA_F16(afr, bfr, vacc);
      }
      __syncthreads();
    }
    const int br = (lane >> 4) * 8;
#pragma unroll
    for (int r = 0; r < 8; ++r)
      shVP[wv * 256 + (br + r) * 16 + m] = vacc[r];
  }
  __syncthreads();

  // --------------------------------------------------------------------
  // Stage 8: reduce K-split partials, bias + relu -> shVR [16][32]
  // --------------------------------------------------------------------
#pragma unroll
  for (int t = 0; t < 2; ++t) {
    int idx = tid + t * 256;
    int b = idx >> 5, e = idx & 31;
    int ntile = e >> 4, col = e & 15;
    float s = v1_b[e];
#pragma unroll
    for (int kq = 0; kq < 4; ++kq)
      s += shVP[(kq * 2 + ntile) * 256 + b * 16 + col];
    shVR[b * 32 + e] = fmaxf(s, 0.f);
  }
  __syncthreads();

  // --------------------------------------------------------------------
  // Stage 9: y[b] = sum_e mix[b][e] + (relu_out[b] @ v2_W + v2_b)
  // --------------------------------------------------------------------
  if (tid < BTILE) {
    int b = tid;
    float v = v2_b[0];
#pragma unroll
    for (int e = 0; e < 32; ++e) v += shVR[b * 32 + e] * v2_W[e];
    float y = v;
#pragma unroll
    for (int e = 0; e < 32; ++e) y += shMix[b * 32 + e];
    out[b0 + b] = y;
  }
}

extern "C" void kernel_launch(void* const* d_in, const int* in_sizes, int n_in,
                              void* d_out, int out_size, void* d_ws, size_t ws_size,
                              hipStream_t stream) {
  (void)in_sizes; (void)n_in; (void)d_ws; (void)ws_size; (void)out_size;
  const float* agent_qs = (const float*)d_in[0];
  const float* states   = (const float*)d_in[1];
  const float* obs      = (const float*)d_in[2];
  const float* gat_W    = (const float*)d_in[3];
  const float* a_src    = (const float*)d_in[4];
  const float* a_dst    = (const float*)d_in[5];
  const float* gat_b    = (const float*)d_in[6];
  const float* w1_W     = (const float*)d_in[7];
  const float* w1_b     = (const float*)d_in[8];
  const float* hb_W     = (const float*)d_in[9];
  const float* hb_b     = (const float*)d_in[10];
  const float* wf_W     = (const float*)d_in[11];
  const float* wf_b     = (const float*)d_in[12];
  const float* v1_W     = (const float*)d_in[13];
  const float* v1_b     = (const float*)d_in[14];
  const float* v2_W     = (const float*)d_in[15];
  const float* v2_b     = (const float*)d_in[16];
  float* out = (float*)d_out;

  (void)hipFuncSetAttribute((const void*)qmixer_fused,
                            hipFuncAttributeMaxDynamicSharedMemorySize, SMEM_BYTES);
  dim3 grid(NBATCH / BTILE), block(256);
  qmixer_fused<<<grid, block, SMEM_BYTES, stream>>>(
      agent_qs, states, obs, gat_W, a_src, a_dst, gat_b,
      w1_W, w1_b, hb_W, hb_b, wf_W, wf_b, v1_W, v1_b, v2_W, v2_b, out);
}
